// EdgeLevelAttentionLayer_65910568124772
// MI455X (gfx1250) — matrix operand order repaired and checked by reading, not compiled, not run on hardware
//
#include <hip/hip_runtime.h>
#include <hip/hip_bf16.h>

// ---------------------------------------------------------------------------
// EdgeLevelAttentionLayer for MI455X (gfx1250, wave32, WMMA).
//
// - Stream each 256MB adjacency ONCE with non-temporal loads (HBM floor
//   ~22us @ 23.3 TB/s) so the 2MB bf16 feature arrays stay L2-resident.
// - Rank-1 masked softmax fused into the aggregation GEMM via
//   V_WMMA_F32_16X16X32_BF16 (transposed product: mask loads coalesce along
//   the edge index, softmax normalizer becomes a per-lane scalar).
// - Split-K 2-way per 16-row tile (2048 waves) + register double-buffered
//   mask loads + global_prefetch_b8 lookahead for HBM latency hiding.
// ---------------------------------------------------------------------------

typedef __bf16 bf16_t;
typedef __attribute__((ext_vector_type(16))) __bf16 v16bf;
typedef __attribute__((ext_vector_type(8)))  __bf16 v8bf;
typedef __attribute__((ext_vector_type(8)))  float  v8f;

#define NROWS 8192   // N == E == 8192
#define DF    64     // feature / embed dim

__device__ __forceinline__ float leaky1(float x) {
    return x >= 0.0f ? x : 0.01f * x;
}

// --- Kernel 1: fold projection matrices into four 64-vectors ---------------
// q0 = W_node @ p_node[64:128], q1 = W_edge @ p_node[0:64],
// q2 = W_edge @ p_edge[0:64],   q3 = W_edge @ p_edge[64:128]
__global__ __launch_bounds__(64) void qprep_kernel(
    const float* __restrict__ Wn, const float* __restrict__ We,
    const float* __restrict__ pn, const float* __restrict__ pe,
    float* __restrict__ q)
{
    int d = threadIdx.x;
    float s0 = 0.f, s1 = 0.f, s2 = 0.f, s3 = 0.f;
    for (int e = 0; e < DF; ++e) {
        float wn = Wn[d * DF + e];
        float we = We[d * DF + e];
        s0 += wn * pn[DF + e];
        s1 += we * pn[e];
        s2 += we * pe[e];
        s3 += we * pe[DF + e];
    }
    q[d] = s0; q[64 + d] = s1; q[128 + d] = s2; q[192 + d] = s3;
}

// --- Kernel 2: per-row dot products (up to 3 score vectors) ----------------
__global__ __launch_bounds__(256) void rowdots_kernel(
    const float* __restrict__ X, const float* __restrict__ q, int nq,
    float* __restrict__ o0, float* __restrict__ o1, float* __restrict__ o2)
{
    int lane = threadIdx.x & 31;
    int wave = threadIdx.x >> 5;
    int row  = blockIdx.x * 8 + wave;
    const float* x = X + (size_t)row * DF;
    float x0 = x[lane];
    float x1 = x[lane + 32];

    float p0 = x0 * q[lane] + x1 * q[lane + 32];
    float p1 = 0.f, p2 = 0.f;
    if (nq > 1) {
        p1 = x0 * q[64 + lane]  + x1 * q[64 + lane + 32];
        p2 = x0 * q[128 + lane] + x1 * q[128 + lane + 32];
    }
    #pragma unroll
    for (int off = 16; off > 0; off >>= 1) {
        p0 += __shfl_xor(p0, off, 32);
        p1 += __shfl_xor(p1, off, 32);
        p2 += __shfl_xor(p2, off, 32);
    }
    if (lane == 0) {
        o0[row] = p0;
        if (nq > 1) { o1[row] = p1; o2[row] = p2; }
    }
}

// --- Kernel 3: global maxes of a_nb and b_nb (softmax stabilizers) ---------
__global__ __launch_bounds__(256) void maxred_kernel(
    const float* __restrict__ a, const float* __restrict__ b,
    float* __restrict__ g)
{
    __shared__ float red[256];
    int tid = threadIdx.x;

    float m = -3.0e38f;
    for (int idx = tid; idx < NROWS; idx += 256) m = fmaxf(m, a[idx]);
    red[tid] = m; __syncthreads();
    for (int s = 128; s > 0; s >>= 1) {
        if (tid < s) red[tid] = fmaxf(red[tid], red[tid + s]);
        __syncthreads();
    }
    if (tid == 0) g[0] = red[0];
    __syncthreads();

    m = -3.0e38f;
    for (int idx = tid; idx < NROWS; idx += 256) m = fmaxf(m, b[idx]);
    red[tid] = m; __syncthreads();
    for (int s = 128; s > 0; s >>= 1) {
        if (tid < s) red[tid] = fmaxf(red[tid], red[tid + s]);
        __syncthreads();
    }
    if (tid == 0) g[1] = red[0];
}

// --- Kernel 4: transpose + convert features to bf16: Xt[c][j] = X[j][c] ----
__global__ __launch_bounds__(256) void transpose_bf16_kernel(
    const float* __restrict__ X, bf16_t* __restrict__ Xt)
{
    int t = blockIdx.x * 256 + threadIdx.x;   // 8192*64 elements
    int j = t >> 6;
    int c = t & 63;
    Xt[(size_t)c * NROWS + j] = (bf16_t)X[t];
}

// --- Kernel 5: fused masked-softmax attention + WMMA aggregation -----------
// Block = 128 threads = 4 waves = 2 row tiles x 2 K-slices.
// D' = A' x B': A' = bf16 feature tile (16 feat rows x 32 neighbors, from the
// pre-transposed Xt so each lane reads two contiguous 16B runs), B' = on-the-
// fly attention-weight tile (32 neighbors x 16 edges; adj[j,i] is contiguous
// in i so the 16 mask dwords per element index coalesce across lanes).
__global__ __launch_bounds__(128) void gat_main_kernel(
    const int*    __restrict__ n2e,   const int*    __restrict__ e2e,
    const float*  __restrict__ a_self, const float* __restrict__ a_nb,
    const float*  __restrict__ b_self, const float* __restrict__ b_nb,
    const bf16_t* __restrict__ Xt_node, const bf16_t* __restrict__ Xt_edge,
    const float*  __restrict__ W_node, const float* __restrict__ W_edge,
    const float*  __restrict__ gmax,
    float* __restrict__ out, int blocksPerSide)
{
    __shared__ float s_anb[NROWS];         // 32 KB: neighbor scores
    __shared__ float s_W[DF * DF];         // 16 KB: output projection
    __shared__ float s_acc[4][DF][16];     // 16 KB: per-wave partial means^T
    __shared__ float s_z[4][16];           // per-wave partial softmax sums
    __shared__ float s_c[4][16];           // per-wave partial neighbor counts

    const int side    = (blockIdx.x >= blocksPerSide) ? 1 : 0;
    const int tileblk = blockIdx.x - side * blocksPerSide;

    const int*    maskp = side ? e2e     : n2e;
    const float*  selfp = side ? b_self  : a_self;
    const float*  nbp   = side ? b_nb    : a_nb;
    const bf16_t* Xt    = side ? Xt_edge : Xt_node;
    const float*  Wp    = side ? W_edge  : W_node;

    const int tid = threadIdx.x;
    for (int idx = tid; idx < NROWS; idx += 128)   s_anb[idx] = nbp[idx];
    for (int idx = tid; idx < DF * DF; idx += 128) s_W[idx]   = Wp[idx];
    __syncthreads();

    const int lane   = tid & 31;
    const int wave   = tid >> 5;
    const int half   = lane >> 4;     // K-sub-slice within the wave
    const int n      = lane & 15;     // edge column within tile
    const int tloc   = wave >> 1;     // local tile (0/1)
    const int kslice = wave & 1;      // K split (0/1)
    const int i0     = (tileblk * 2 + tloc) * 16;
    const int i      = i0 + n;        // this lane's edge row

    const float selfi = selfp[i];
    const float smax  = leaky1(selfi + gmax[side]);   // stable row-max bound

    v8f acc[4];
    #pragma unroll
    for (int t = 0; t < 4; ++t)
        #pragma unroll
        for (int r = 0; r < 8; ++r) acc[t][r] = 0.0f;

    float zacc = 0.0f;
    float cacc = 0.0f;

    const int* mbase = maskp + i;    // column i of adjacency, stride NROWS
    const bf16_t* xrow[4];
    #pragma unroll
    for (int t = 0; t < 4; ++t)
        xrow[t] = Xt + (size_t)(t * 16 + n) * NROWS + 16 * half;

    const int kstart = kslice * (NROWS / 2);
    const int kend   = kstart + (NROWS / 2);

    // preload first mask tile (this lane's 16 (j,i) dwords, NT policy)
    int mv[16];
    #pragma unroll
    for (int e = 0; e < 16; ++e)
        mv[e] = __builtin_nontemporal_load(
                    mbase + (size_t)(kstart + 16 * half + e) * NROWS);

    #pragma unroll 2
    for (int jb = kstart; jb < kend; jb += 32) {
        // prefetch the full tile 4 iterations ahead: lane -> one 64B line
        {
            int jpf = jb + 128 + lane;
            jpf = (jpf < NROWS) ? jpf : (NROWS - 1);
            __builtin_prefetch(
                (const void*)(maskp + (size_t)jpf * NROWS + i0), 0, 0);
        }

        // double-buffer: issue next iteration's mask loads now
        const int jbn = (jb + 32 < kend) ? (jb + 32) : kstart;
        int mvn[16];
        #pragma unroll
        for (int e = 0; e < 16; ++e)
            mvn[e] = __builtin_nontemporal_load(
                         mbase + (size_t)(jbn + 16 * half + e) * NROWS);

        const int jB = jb + 16 * half;

        // neighbor scores from LDS (broadcast within each half-wave)
        float anbv[16];
        const float4* a4 = (const float4*)(&s_anb[jB]);
        #pragma unroll
        for (int qd = 0; qd < 4; ++qd) {
            float4 tv = a4[qd];
            anbv[4 * qd + 0] = tv.x; anbv[4 * qd + 1] = tv.y;
            anbv[4 * qd + 2] = tv.z; anbv[4 * qd + 3] = tv.w;
        }

        // build B' tile: w = mask ? exp(leaky(self+nb) - smax) : 0
        v16bf bv;
        #pragma unroll
        for (int e = 0; e < 16; ++e) {
            float s  = leaky1(selfi + anbv[e]);
            float wm = (mv[e] > 0) ? 1.0f : 0.0f;
            float w  = wm * __expf(s - smax);
            zacc += w;
            cacc += wm;
            bv[e] = (bf16_t)w;
        }

        // 4 feature tiles x same weight tile
        #pragma unroll
        for (int t = 0; t < 4; ++t) {
            const bf16_t* xr = xrow[t] + jb;
            v8bf lo = *(const v8bf*)(xr);        // K = 16*half + 0..7
            v8bf hi = *(const v8bf*)(xr + 8);    // K = 16*half + 8..15
            v16bf av = __builtin_shufflevector(lo, hi,
                0, 1, 2, 3, 4, 5, 6, 7, 8, 9, 10, 11, 12, 13, 14, 15);
            acc[t] = __builtin_amdgcn_wmma_f32_16x16x32_bf16(
                false, av, false, bv, (short)0, acc[t], false, false);
        }

        #pragma unroll
        for (int e = 0; e < 16; ++e) mv[e] = mvn[e];
    }

    // combine the two half-wave K-sub-slices within this wave
    float zW = zacc + __shfl_xor(zacc, 16, 32);
    float cW = cacc + __shfl_xor(cacc, 16, 32);

    // publish this wave's partials (D' element (t,r): c = t*16+r+8*half)
    #pragma unroll
    for (int t = 0; t < 4; ++t)
        #pragma unroll
        for (int r = 0; r < 8; ++r)
            s_acc[wave][t * 16 + r + 8 * half][n] = acc[t][r];
    if (half == 0) { s_z[wave][n] = zW; s_c[wave][n] = cW; }
    __syncthreads();

    if (wave & 1) return;   // odd K-slice waves are done

    // combine the two K-slices and normalize: means = acc_total/(z*cnt)
    float zT  = s_z[wave][n] + s_z[wave + 1][n];
    float cT  = s_c[wave][n] + s_c[wave + 1][n];
    float inv = __builtin_amdgcn_rcpf(zT * cT);

    #pragma unroll 8
    for (int c0 = 0; c0 < 32; ++c0) {
        int c = half * 32 + c0;
        s_acc[wave][c][n] =
            (s_acc[wave][c][n] + s_acc[wave + 1][c][n]) * inv;
    }
    asm volatile("s_wait_dscnt 0" ::: "memory");  // wave-local LDS RAW

    // epilogue: out[i, :] = leaky(means[i, :] @ W); each lane does 32 cols
    float o[32];
    #pragma unroll
    for (int k = 0; k < 32; ++k) o[k] = 0.0f;
    for (int f = 0; f < DF; ++f) {
        float m = s_acc[wave][f][n];
        const float4* w4 = (const float4*)(&s_W[f * DF + half * 32]);
        #pragma unroll
        for (int qd = 0; qd < 8; ++qd) {
            float4 wv = w4[qd];
            o[4 * qd + 0] += m * wv.x; o[4 * qd + 1] += m * wv.y;
            o[4 * qd + 2] += m * wv.z; o[4 * qd + 3] += m * wv.w;
        }
    }
    float* orow = out + (size_t)i * (2 * DF) + side * DF + half * 32;
    #pragma unroll
    for (int k = 0; k < 32; ++k)
        __builtin_nontemporal_store(leaky1(o[k]), orow + k);
}

// ---------------------------------------------------------------------------
extern "C" void kernel_launch(void* const* d_in, const int* in_sizes, int n_in,
                              void* d_out, int out_size, void* d_ws, size_t ws_size,
                              hipStream_t stream) {
    (void)in_sizes; (void)n_in; (void)out_size; (void)ws_size;

    const float* node_features = (const float*)d_in[0];
    const float* edge_features = (const float*)d_in[1];
    const int*   e2e           = (const int*)  d_in[2];
    const int*   n2e           = (const int*)  d_in[3];
    const float* W_node        = (const float*)d_in[4];
    const float* W_edge        = (const float*)d_in[5];
    const float* p_node        = (const float*)d_in[6];
    const float* p_edge        = (const float*)d_in[7];

    // workspace layout (floats)
    float* ws      = (float*)d_ws;
    float* q       = ws;                  // 256
    float* a_nb    = ws + 256;            // 8192
    float* a_self  = a_nb  + NROWS;       // 8192
    float* b_self  = a_self + NROWS;      // 8192
    float* b_nb    = b_self + NROWS;      // 8192
    float* gmax    = b_nb  + NROWS;       // 2 (+30 pad)
    bf16_t* Xt_node = (bf16_t*)(gmax + 32);         // 8192*64 bf16 (1 MB)
    bf16_t* Xt_edge = Xt_node + (size_t)NROWS * DF; // 1 MB

    qprep_kernel<<<1, 64, 0, stream>>>(W_node, W_edge, p_node, p_edge, q);

    rowdots_kernel<<<NROWS / 8, 256, 0, stream>>>(
        node_features, q, 1, a_nb, nullptr, nullptr);
    rowdots_kernel<<<NROWS / 8, 256, 0, stream>>>(
        edge_features, q + 64, 3, a_self, b_self, b_nb);

    maxred_kernel<<<1, 256, 0, stream>>>(a_nb, b_nb, gmax);

    transpose_bf16_kernel<<<(NROWS * DF) / 256, 256, 0, stream>>>(
        node_features, Xt_node);
    transpose_bf16_kernel<<<(NROWS * DF) / 256, 256, 0, stream>>>(
        edge_features, Xt_edge);

    const int blocksPerSide = NROWS / (16 * 2);   // 2 tiles per block -> 256
    gat_main_kernel<<<2 * blocksPerSide, 128, 0, stream>>>(
        n2e, e2e, a_self, a_nb, b_self, b_nb,
        Xt_node, Xt_edge, W_node, W_edge, gmax,
        (float*)d_out, blocksPerSide);
}
